// TimeMixing_19207093747760
// MI455X (gfx1250) — matrix operand order repaired and checked by reading, not compiled, not run on hardware
//
#include <hip/hip_runtime.h>
#include <hip/hip_bf16.h>

// ---------------------------------------------------------------------------
// Fused GRU-gate batched GEMM for MI455X (gfx1250, wave32, WMMA).
//   gi = einsum('nfs,fgs->nfg', x, W_ih) + b_ih        (64 GEMMs 2048x512x1536)
//   h  = (1-sigmoid(gz)) * tanh(gn_pre + sigmoid(gr)*b_hn)   (fused epilogue)
// Precision: bf16 hi/lo split (Ah*Bh + Ah*Bl + Al*Bh), fp32 accumulation
// via v_wmma_f32_16x16x32_bf16. Split happens ONCE at staging (global->reg->
// cvt->LDS as bf16), so per-wave fragment builds are pure ds_load_b128.
// Register double-buffer: next tile's global loads issue before WMMA section.
// ---------------------------------------------------------------------------

typedef __attribute__((ext_vector_type(16))) __bf16 v16bf;
typedef __attribute__((ext_vector_type(8)))  __bf16 v8bf;
typedef __attribute__((ext_vector_type(4)))  __bf16 v4bf;
typedef __attribute__((ext_vector_type(8)))  float  v8f;

constexpr int N_ = 2048;
constexpr int F_ = 64;
constexpr int S_ = 512;
constexpr int D_ = 512;
constexpr int G_ = 3 * D_;   // 1536

constexpr int BM = 128;      // rows per block
constexpr int BN = 64;       // h-columns per block (x3 gates internally)
constexpr int KT = 32;       // WMMA K chunk (bf16)
constexpr int PB = KT + 8;   // LDS bf16 row pitch (80 B: 16B-aligned, 20-bank stride)

__device__ __forceinline__ v8f wmma_bf16(v16bf a, v16bf b, v8f c) {
  return __builtin_amdgcn_wmma_f32_16x16x32_bf16(
      /*neg_a=*/false, a, /*neg_b=*/false, b,
      /*c_mod=*/(short)0, c, /*reuse_a=*/false, /*reuse_b=*/false);
}

// Split 4 fp32 -> packed bf16 hi + bf16 lo (Ootomo split).
__device__ __forceinline__ void split4(const float4& v, v4bf& h, v4bf& l) {
  const float* a = &v.x;
#pragma unroll
  for (int t = 0; t < 4; ++t) {
    __bf16 hh = (__bf16)a[t];
    h[t] = hh;
    l[t] = (__bf16)(a[t] - (float)hh);
  }
}

__global__ __launch_bounds__(256, 1)
void gru_gates_wmma_kernel(const float* __restrict__ x,
                           const float* __restrict__ W,
                           const float* __restrict__ bih,
                           const float* __restrict__ bhh,
                           float* __restrict__ out) {
  __shared__ __bf16 shAh[BM * PB];          // 10240 B each
  __shared__ __bf16 shAl[BM * PB];
  __shared__ __bf16 shBh[3 * BN * PB];      // 15360 B each
  __shared__ __bf16 shBl[3 * BN * PB];

  const int tid   = threadIdx.x;
  const int lane  = tid & 31;
  const int wave  = tid >> 5;        // 0..7
  const int wm    = wave & 3;        // M-wave: rows wm*32 .. +31
  const int wn    = wave >> 2;       // N-wave: cols wn*32 .. +31

  const int nbase = blockIdx.x * BN;     // 0..448
  const int mbase = blockIdx.y * BM;     // 0..1920
  const int f     = blockIdx.z;          // 0..63

  const int lrow  = lane & 15;
  const int khalf = lane >> 4;           // 0 or 1

  // Per-thread staging slices (constant across K-steps except the K offset).
  const int arow = tid >> 1;             // 0..127  (A: 128 rows x 2 quads)
  const int aq   = (tid & 1) * 16;       // float offset within K-chunk half? no:
  // A: 128 rows * 32 K = 4096 floats = 1024 float4; 256 thr -> 4 each.
  // job = tid + p*256 : row = job>>3 (0..127), q = job&7.
  (void)arow; (void)aq;

  float4 ra[4], rb[6];

  auto load_tiles = [&](int ks) {
#pragma unroll
    for (int p = 0; p < 4; ++p) {
      int job = tid + p * 256;
      int row = job >> 3, q = job & 7;
      ra[p] = *reinterpret_cast<const float4*>(
          x + ((size_t)(mbase + row) * F_ + f) * S_ + ks + q * 4);
    }
#pragma unroll
    for (int p = 0; p < 6; ++p) {
      int job = tid + p * 256;
      int row = job >> 3, q = job & 7;       // row = gate*64 + c
      rb[p] = *reinterpret_cast<const float4*>(
          W + ((size_t)f * G_ + (row >> 6) * D_ + nbase + (row & 63)) * S_ + ks + q * 4);
    }
  };

  auto store_tiles = [&]() {
#pragma unroll
    for (int p = 0; p < 4; ++p) {
      int job = tid + p * 256;
      int row = job >> 3, q = job & 7;
      v4bf h, l;
      split4(ra[p], h, l);
      *reinterpret_cast<v4bf*>(&shAh[row * PB + q * 4]) = h;
      *reinterpret_cast<v4bf*>(&shAl[row * PB + q * 4]) = l;
    }
#pragma unroll
    for (int p = 0; p < 6; ++p) {
      int job = tid + p * 256;
      int row = job >> 3, q = job & 7;
      v4bf h, l;
      split4(rb[p], h, l);
      *reinterpret_cast<v4bf*>(&shBh[row * PB + q * 4]) = h;
      *reinterpret_cast<v4bf*>(&shBl[row * PB + q * 4]) = l;
    }
  };

  v8f acc[3][2][2];                      // [gate][mfrag][nfrag]
#pragma unroll
  for (int g = 0; g < 3; ++g)
#pragma unroll
    for (int i = 0; i < 2; ++i)
#pragma unroll
      for (int j = 0; j < 2; ++j)
#pragma unroll
        for (int r = 0; r < 8; ++r) acc[g][i][j][r] = 0.0f;

  load_tiles(0);

  for (int ks = 0; ks < S_; ks += KT) {
    __syncthreads();                  // previous iteration done reading LDS
    store_tiles();
    __syncthreads();

    if (ks + KT < S_) load_tiles(ks + KT);          // overlap with WMMAs
    if (ks + 2 * KT < S_) {                         // L2 prefetch 2 steps ahead
      __builtin_prefetch(x + ((size_t)(mbase + (tid >> 1)) * F_ + f) * S_
                           + ks + 2 * KT + (tid & 1) * 16, 0, 1);
      __builtin_prefetch(W + ((size_t)f * G_ + ((tid * 6) >> 3)) * S_
                           + ks + 2 * KT, 0, 1);
    }

    // ---- A fragments (16-bit A 16x32 layout, ISA 7.12.2) ----
    // lanes 0-15: M=lrow, elems0-7 = K 0..7,  elems8-15 = K 16..23
    // lanes16-31: M=lrow, elems0-7 = K 8..15, elems8-15 = K 24..31
    v16bf ahi[2], alo[2];
#pragma unroll
    for (int i = 0; i < 2; ++i) {
      const __bf16* ph = &shAh[(wm * 32 + i * 16 + lrow) * PB + khalf * 8];
      const __bf16* pl = &shAl[(wm * 32 + i * 16 + lrow) * PB + khalf * 8];
      v8bf h0 = *reinterpret_cast<const v8bf*>(ph);
      v8bf h1 = *reinterpret_cast<const v8bf*>(ph + 16);
      v8bf l0 = *reinterpret_cast<const v8bf*>(pl);
      v8bf l1 = *reinterpret_cast<const v8bf*>(pl + 16);
      ahi[i] = __builtin_shufflevector(h0, h1, 0,1,2,3,4,5,6,7,8,9,10,11,12,13,14,15);
      alo[i] = __builtin_shufflevector(l0, l1, 0,1,2,3,4,5,6,7,8,9,10,11,12,13,14,15);
    }

    // ---- per gate: B fragments (lane = N col; K contiguous) + WMMAs ----
#pragma unroll
    for (int g = 0; g < 3; ++g) {
      v16bf bhi[2], blo[2];
#pragma unroll
      for (int j = 0; j < 2; ++j) {
        const int brow = (g * BN + wn * 32 + j * 16 + lrow) * PB + khalf * 8;
        v8bf h0 = *reinterpret_cast<const v8bf*>(&shBh[brow]);
        v8bf h1 = *reinterpret_cast<const v8bf*>(&shBh[brow + 16]);
        v8bf l0 = *reinterpret_cast<const v8bf*>(&shBl[brow]);
        v8bf l1 = *reinterpret_cast<const v8bf*>(&shBl[brow + 16]);
        bhi[j] = __builtin_shufflevector(h0, h1, 0,1,2,3,4,5,6,7,8,9,10,11,12,13,14,15);
        blo[j] = __builtin_shufflevector(l0, l1, 0,1,2,3,4,5,6,7,8,9,10,11,12,13,14,15);
      }
#pragma unroll
      for (int i = 0; i < 2; ++i)
#pragma unroll
        for (int j = 0; j < 2; ++j) {
          acc[g][i][j] = wmma_bf16(ahi[i], bhi[j], acc[g][i][j]); // hi*hi
          acc[g][i][j] = wmma_bf16(ahi[i], blo[j], acc[g][i][j]); // hi*lo
          acc[g][i][j] = wmma_bf16(alo[i], bhi[j], acc[g][i][j]); // lo*hi
        }
    }
  }

  // ---- fused GRU epilogue + store ----
  // C layout: VGPR r -> M = r + 8*(lane>>4); N = lane&15
#pragma unroll
  for (int i = 0; i < 2; ++i)
#pragma unroll
    for (int j = 0; j < 2; ++j) {
      const int col = nbase + wn * 32 + j * 16 + (lane & 15);
      const size_t bo = (size_t)f * G_ + col;
      const float bir = bih[bo];            const float bhr = bhh[bo];
      const float biz = bih[bo + D_];       const float bhz = bhh[bo + D_];
      const float bin = bih[bo + 2 * D_];   const float bhn = bhh[bo + 2 * D_];
#pragma unroll
      for (int r = 0; r < 8; ++r) {
        const int row = mbase + wm * 32 + i * 16 + r + ((lane >> 4) << 3);
        const float ir = acc[0][i][j][r] + bir + bhr;
        const float iz = acc[1][i][j][r] + biz + bhz;
        const float in = acc[2][i][j][r] + bin;
        const float rg = 1.0f / (1.0f + __expf(-ir));
        const float zg = 1.0f / (1.0f + __expf(-iz));
        const float ng = tanhf(in + rg * bhn);
        out[((size_t)row * F_ + f) * D_ + col] = (1.0f - zg) * ng;
      }
    }
}

extern "C" void kernel_launch(void* const* d_in, const int* in_sizes, int n_in,
                              void* d_out, int out_size, void* d_ws, size_t ws_size,
                              hipStream_t stream) {
  const float* x   = (const float*)d_in[0];
  const float* W   = (const float*)d_in[1];
  const float* bih = (const float*)d_in[2];
  const float* bhh = (const float*)d_in[3];
  float* out = (float*)d_out;

  dim3 grid(D_ / BN, N_ / BM, F_);   // (8, 16, 64)
  dim3 block(256);
  gru_gates_wmma_kernel<<<grid, block, 0, stream>>>(x, W, bih, bhh, out);
}
